// MixtralBLockSparseTop2MLP_HQQ_80169859547187
// MI455X (gfx1250) — matrix operand order repaired and checked by reading, not compiled, not run on hardware
//
#include <hip/hip_runtime.h>
#include <cstdint>

#define DEVINL __device__ __forceinline__

typedef __attribute__((ext_vector_type(16))) __bf16 v16bf;
typedef __attribute__((ext_vector_type(8)))  float  v8f;
typedef __attribute__((ext_vector_type(4)))  int    v4i;

static constexpr int Tdim  = 512;
static constexpr int Hdim  = 4096;
static constexpr int Fdim  = 14336;
static constexpr int GROUP = 64;
static constexpr int LDSP  = 72;   // padded LDS row stride (144 B = 9*16 B)

// native float -> bf16 (RNE)
DEVINL unsigned short f2bf(float f) {
  __bf16 b = (__bf16)f;
  return __builtin_bit_cast(unsigned short, b);
}

DEVINL unsigned int pack2bf(float lo, float hi) {
  return (unsigned int)f2bf(lo) | ((unsigned int)f2bf(hi) << 16);
}

union BF16x16 {
  v16bf v;
  uint4 q[2];
};

// ---------------- kernel 0: f32 -> bf16 convert ----------------
__global__ __launch_bounds__(256)
void f32_to_bf16_kernel(const float* __restrict__ in,
                        unsigned short* __restrict__ out, int n) {
  int i = blockIdx.x * blockDim.x + threadIdx.x;
  if (i < n) out[i] = f2bf(in[i]);
}

// Dequantize 8 preloaded codes -> 8 packed bf16 -> one ds_store_b128.
DEVINL void dequant8_store(unsigned short* __restrict__ dst,
                           v4i qa, v4i qb, float s, float z) {
  const float nzs = -z * s;
  int qv[8] = {qa.x, qa.y, qa.z, qa.w, qb.x, qb.y, qb.z, qb.w};
  uint4 o;
  unsigned int r[4];
#pragma unroll
  for (int e = 0; e < 4; ++e) {
    r[e] = pack2bf(fmaf((float)qv[2 * e + 0], s, nzs),
                   fmaf((float)qv[2 * e + 1], s, nzs));
  }
  o.x = r[0]; o.y = r[1]; o.z = r[2]; o.w = r[3];
  *(uint4*)dst = o;
}

DEVINL v8f wmma_bf16(v16bf a, v16bf b, v8f c) {
  return __builtin_amdgcn_wmma_f32_16x16x32_bf16(
      false, a, false, b, (short)0, c, false, false);
}

// ---------------- WMMA GEMM: C[M,N] = A[M,K] * dequant(Wq[N,K])^T ----------
// Block: 256 threads = 8 waves. Block tile N=32, M=64*MT.
// Per 64-K chunk (== one quant group):
//   stage:   A slice (MT*64 x 64 bf16) cooperatively copied L2 -> LDS
//            (one exposed L2 latency per chunk, not per fragment);
//            weight codes (preloaded into regs last chunk) dequantized ONCE
//            per element per block and ds_store'd as bf16.
//   sync;  issue next chunk's NT weight loads (hidden under WMMAs);
//   compute: A and B fragments via ds_load_b128 (short-latency DScnt waits),
//            2 k-steps x MT tiles x {W1,W3} WMMAs; sync.
template <bool FUSED, int MT>
__global__ __launch_bounds__(256)
void hqq_wmma_gemm(const unsigned short* __restrict__ A,  // bf16 [512 x K]
                   const int*   __restrict__ Wq1,
                   const float* __restrict__ sc1,
                   const float* __restrict__ zp1,
                   const int*   __restrict__ Wq3,
                   const float* __restrict__ sc3,
                   const float* __restrict__ zp3,
                   unsigned short* __restrict__ out_bf16,  // used if FUSED
                   float* __restrict__ out_f32,            // used if !FUSED
                   int N, int K) {
  constexpr int ROWS = MT * 64;           // block M extent
  __shared__ __align__(16) unsigned short lds1[32 * LDSP];
  __shared__ __align__(16) unsigned short lds3[32 * LDSP];
  __shared__ __align__(16) unsigned short ldsA[ROWS * LDSP];

  const int lane = threadIdx.x & 31;
  const int wid  = threadIdx.x >> 5;
  const int half = lane >> 4;   // 0: lanes 0-15, 1: lanes 16-31
  const int mr   = lane & 15;

  const int nblk = blockIdx.x * 32;
  const int ntr  = (wid & 1) * 16 + mr;            // B tile row (0..31)
  const int mloc = (wid >> 1) * (MT * 16);         // block-local M base

  // cooperative weight-stage coordinates: thread -> (weight row, 8 codes)
  const int tn = threadIdx.x >> 3;        // 0..31
  const int tk = (threadIdx.x & 7) * 8;   // 0..56
  const int  gpr   = K / GROUP;
  const long wrowT = (long)(nblk + tn) * K + tk;
  const long growT = (long)(nblk + tn) * gpr;

  v8f acc1[MT];
  v8f acc3[MT];
#pragma unroll
  for (int i = 0; i < MT; ++i) {
    acc1[i] = (v8f){};
    if constexpr (FUSED) acc3[i] = (v8f){};
  }

  const int nchunk = K / 64;

  // ---- preload chunk 0 weight codes + scale/zero into registers ----
  v4i p1a, p1b, p3a, p3b;
  float s1v, z1v, s3v, z3v;
  {
    const v4i* vp = (const v4i*)(Wq1 + wrowT);
    p1a = __builtin_nontemporal_load(vp + 0);
    p1b = __builtin_nontemporal_load(vp + 1);
    s1v = sc1[growT]; z1v = zp1[growT];
    if constexpr (FUSED) {
      const v4i* vq = (const v4i*)(Wq3 + wrowT);
      p3a = __builtin_nontemporal_load(vq + 0);
      p3b = __builtin_nontemporal_load(vq + 1);
      s3v = sc3[growT]; z3v = zp3[growT];
    }
  }

  for (int c = 0; c < nchunk; ++c) {
    // ---- stage A chunk: L2 -> LDS, 64 cols x ROWS rows of bf16 ----
    // thread handles ROWS/256 full rows, 128 B each (8 b128), in 2 halves
#pragma unroll
    for (int rr = 0; rr < ROWS / 256; ++rr) {
      const int row = (int)threadIdx.x + rr * 256;
      const uint4* src =
          (const uint4*)(A + (long)(blockIdx.y * ROWS + row) * K + c * 64);
      uint4* dst = (uint4*)&ldsA[row * LDSP];
      uint4 t0 = src[0], t1 = src[1], t2 = src[2], t3 = src[3];
      dst[0] = t0; dst[1] = t1; dst[2] = t2; dst[3] = t3;
      uint4 t4 = src[4], t5 = src[5], t6 = src[6], t7 = src[7];
      dst[4] = t4; dst[5] = t5; dst[6] = t6; dst[7] = t7;
    }

    // ---- dequant preloaded weight chunk into LDS ----
    dequant8_store(&lds1[tn * LDSP + tk], p1a, p1b, s1v, z1v);
    if constexpr (FUSED)
      dequant8_store(&lds3[tn * LDSP + tk], p3a, p3b, s3v, z3v);
    __syncthreads();

    // ---- issue next chunk's global NT weight loads (hidden by WMMAs) ----
    if (c + 1 < nchunk) {
      const v4i* vp = (const v4i*)(Wq1 + wrowT + (long)(c + 1) * 64);
      p1a = __builtin_nontemporal_load(vp + 0);
      p1b = __builtin_nontemporal_load(vp + 1);
      s1v = sc1[growT + c + 1]; z1v = zp1[growT + c + 1];
      if constexpr (FUSED) {
        const v4i* vq = (const v4i*)(Wq3 + wrowT + (long)(c + 1) * 64);
        p3a = __builtin_nontemporal_load(vq + 0);
        p3b = __builtin_nontemporal_load(vq + 1);
        s3v = sc3[growT + c + 1]; z3v = zp3[growT + c + 1];
      }
    }

    // ---- compute: two 32-K WMMA steps per chunk, all operands from LDS ----
#pragma unroll
    for (int k0l = 0; k0l < 64; k0l += 32) {
      // B fragment: lane = column ntr, elems e -> K = k0l + 16*half + e
      BF16x16 b1, b3;
      {
        const unsigned short* bp = &lds1[ntr * LDSP + k0l + 16 * half];
        b1.q[0] = *(const uint4*)bp;
        b1.q[1] = *(const uint4*)(bp + 8);
      }
      if constexpr (FUSED) {
        const unsigned short* bp = &lds3[ntr * LDSP + k0l + 16 * half];
        b3.q[0] = *(const uint4*)bp;
        b3.q[1] = *(const uint4*)(bp + 8);
      }
      // A fragment, same contiguous k order as B: elems 0-15 = K[16h..+15]
#pragma unroll
      for (int i = 0; i < MT; ++i) {
        const unsigned short* Ar =
            &ldsA[(mloc + i * 16 + mr) * LDSP + k0l + 16 * half];
        BF16x16 a;
        a.q[0] = *(const uint4*)Ar;
        a.q[1] = *(const uint4*)(Ar + 8);
        acc1[i] = wmma_bf16(a.v, b1.v, acc1[i]);
        if constexpr (FUSED) acc3[i] = wmma_bf16(a.v, b3.v, acc3[i]);
      }
    }
    __syncthreads();
  }

  // ---- epilogue: C layout elem r -> (M = r + 8*half, N = mr) ----
  const int ncol = nblk + ntr;
#pragma unroll
  for (int i = 0; i < MT; ++i) {
#pragma unroll
    for (int r = 0; r < 8; ++r) {
      const int row = blockIdx.y * ROWS + mloc + i * 16 + r + 8 * half;
      if constexpr (FUSED) {
        float h1  = acc1[i][r];
        float h3  = acc3[i][r];
        float sig = __builtin_amdgcn_rcpf(1.0f + __expf(-h1));  // fast sigmoid
        out_bf16[(long)row * N + ncol] = f2bf(h1 * sig * h3);
      } else {
        out_f32[(long)row * N + ncol] = acc1[i][r];
      }
    }
  }
}

extern "C" void kernel_launch(void* const* d_in, const int* in_sizes, int n_in,
                              void* d_out, int out_size, void* d_ws, size_t ws_size,
                              hipStream_t stream) {
  const float* x    = (const float*)d_in[0];
  const int*   W1q  = (const int*)  d_in[1];
  const float* s1   = (const float*)d_in[2];
  const float* z1   = (const float*)d_in[3];
  const int*   W3q  = (const int*)  d_in[4];
  const float* s3   = (const float*)d_in[5];
  const float* z3   = (const float*)d_in[6];
  const int*   W2q  = (const int*)  d_in[7];
  const float* s2   = (const float*)d_in[8];
  const float* z2   = (const float*)d_in[9];
  float* out = (float*)d_out;

  // workspace: x_bf16 [T*H] (4 MB) | g_bf16 [T*F] (14.7 MB) -- L2-resident
  unsigned short* x16 = (unsigned short*)d_ws;
  unsigned short* g16 = x16 + (size_t)Tdim * Hdim;

  // 0) convert activations to bf16
  {
    int n = Tdim * Hdim;
    f32_to_bf16_kernel<<<(n + 255) / 256, 256, 0, stream>>>(x, x16, n);
  }

  // 1) g = silu(x W1^T) * (x W3^T) -> bf16 [T, F]
  //    MT=8: block spans all 512 rows -> W1/W3 streamed from HBM once.
  {
    dim3 grid(Fdim / 32, 1);
    hqq_wmma_gemm<true, 8><<<grid, 256, 0, stream>>>(
        x16, W1q, s1, z1, W3q, s3, z3, g16, nullptr, Fdim, Hdim);
  }

  // 2) out = g W2^T -> f32 [T, H]
  //    MT=4, grid.y=2: W2 streamed twice, 256 blocks of parallelism.
  {
    dim3 grid(Hdim / 32, 2);
    hqq_wmma_gemm<false, 4><<<grid, 256, 0, stream>>>(
        g16, W2q, s2, z2, nullptr, nullptr, nullptr, nullptr, out, Hdim, Fdim);
  }
}
